// Attention_60112362275620
// MI455X (gfx1250) — matrix operand order repaired and checked by reading, not compiled, not run on hardware
//
#include <hip/hip_runtime.h>

// ---------------------------------------------------------------------------
// MI455X (gfx1250, wave32) fused attention:
//   1) fp32 -> bf16 staging (x, wq, wk, wv, wo)
//   2) fused QKV projection GEMM + RoPE (+ 1/sqrt(HD) baked into Q), WMMA bf16
//   3) flash attention (online softmax), WMMA bf16 for QK^T and P@V
//   4) output projection GEMM, WMMA bf16, fp32 output
// All matmuls use v_wmma_f32_16x16x32_bf16. One wave (32 threads) per block
// so EXEC is all ones at every WMMA. GEMM waves compute 32x64 tiles (2x B
// reuse); flash waves process 32 queries (2x K/V reuse).
// ---------------------------------------------------------------------------

typedef __bf16 bf16;
typedef __attribute__((ext_vector_type(16))) __bf16 v16bf;
typedef __attribute__((ext_vector_type(8)))  __bf16 v8bf;
typedef __attribute__((ext_vector_type(4)))  __bf16 v4bf;
typedef __attribute__((ext_vector_type(8)))  float  v8f;
typedef __attribute__((ext_vector_type(4)))  float  v4f;

#define B_   2
#define S_   2048
#define D_   2048
#define H_   32
#define KVH_ 8
#define HD_  64

#define WMMA_BF16(a, b, c) \
  __builtin_amdgcn_wmma_f32_16x16x32_bf16(false, (a), false, (b), (short)0, (c), false, false)

static __device__ __forceinline__ v8f zero8() {
  v8f z = {0.f, 0.f, 0.f, 0.f, 0.f, 0.f, 0.f, 0.f};
  return z;
}
static __device__ __forceinline__ v8bf ld8(const bf16* p) { return *(const v8bf*)p; }
static __device__ __forceinline__ v16bf cat16(v8bf a, v8bf b) {
  return __builtin_shufflevector(a, b, 0,1,2,3,4,5,6,7,8,9,10,11,12,13,14,15);
}

// A fragment (16x32, M x K), row-major source with row stride ld (elements).
// Lanes 0-15 hold row=lane, K in {0..7,16..23}; lanes 16-31 hold row=lane-16,
// K in {8..15,24..31}. Two contiguous 16B loads per lane.
static __device__ __forceinline__ v16bf load_a(const bf16* src, int ld, int lane) {
  const int r  = lane & 15;
  const int hi = lane >> 4;
  const bf16* base = src + (size_t)r * ld + hi * 8;
  return cat16(ld8(base), ld8(base + 16));
}

// B fragment given B^T rows (N rows of length K, contiguous in K), stride ld.
// Lanes 0-15: column n=lane, K in {0..7,16..23}; lanes 16-31: column n=lane-16, K in {8..15,24..31}.
static __device__ __forceinline__ v16bf load_bt(const bf16* src, int ld, int lane) {
  const int n  = lane & 15;
  const int hi = lane >> 4;
  const bf16* base = src + (size_t)n * ld + hi * 8;
  return cat16(ld8(base), ld8(base + 16));
}

// Row-wise reductions over the 16 key-columns of one D tile: masks 1,2,4,8
// stay inside each 16-lane half, matching the C/D row mapping.
static __device__ __forceinline__ float rowmax16(float v) {
  v = fmaxf(v, __shfl_xor(v, 1, 32));
  v = fmaxf(v, __shfl_xor(v, 2, 32));
  v = fmaxf(v, __shfl_xor(v, 4, 32));
  v = fmaxf(v, __shfl_xor(v, 8, 32));
  return v;
}
static __device__ __forceinline__ float rowsum16(float v) {
  v += __shfl_xor(v, 1, 32);
  v += __shfl_xor(v, 2, 32);
  v += __shfl_xor(v, 4, 32);
  v += __shfl_xor(v, 8, 32);
  return v;
}

// ---------------------------------------------------------------------------
// Kernel 1: fp32 -> bf16 conversion, 4 elements/thread.
// ---------------------------------------------------------------------------
__global__ __launch_bounds__(256) void cvt_kernel(const float* __restrict__ in,
                                                  bf16* __restrict__ out, int n4) {
  int i = blockIdx.x * 256 + threadIdx.x;
  if (i < n4) {
    v4f x = ((const v4f*)in)[i];
    v4bf y;
    #pragma unroll
    for (int j = 0; j < 4; ++j) y[j] = (bf16)x[j];
    ((v4bf*)out)[i] = y;
  }
}

// ---------------------------------------------------------------------------
// Kernel 2: fused QKV GEMM + RoPE.
// grid = (M/32 = 128, 48). Each wave: 32 rows x 64 cols (one head wide).
// N-blocks 0..31 -> Q (rope, *0.125), 32..39 -> K (rope), 40..47 -> V (store
// transposed [b][kvh][hd][s] with contiguous b128 stores).
// ---------------------------------------------------------------------------
__global__ __launch_bounds__(32) void qkv_rope_kernel(
    const bf16* __restrict__ xb,  const bf16* __restrict__ wqb,
    const bf16* __restrict__ wkb, const bf16* __restrict__ wvb,
    bf16* __restrict__ qb, bf16* __restrict__ kbuf, bf16* __restrict__ vtb) {
  const int lane  = threadIdx.x & 31;
  const int m0    = blockIdx.x * 32;
  const int ncol0 = blockIdx.y * 64;

  const bf16* W;
  int region, headStart;
  if (ncol0 < 2048)      { W = wqb; region = 0; headStart = ncol0; }
  else if (ncol0 < 2560) { W = wkb; region = 1; headStart = ncol0 - 2048; }
  else                   { W = wvb; region = 2; headStart = ncol0 - 2560; }

  const bf16* xA = xb + (size_t)m0 * D_;
  v8f acc[2][4];
  #pragma unroll
  for (int mi = 0; mi < 2; ++mi)
    #pragma unroll
    for (int t = 0; t < 4; ++t) acc[mi][t] = zero8();

  for (int k0 = 0; k0 < D_; k0 += 32) {
    v16bf a0 = load_a(xA + k0, D_, lane);
    v16bf a1 = load_a(xA + (size_t)16 * D_ + k0, D_, lane);
    #pragma unroll
    for (int t = 0; t < 4; ++t) {
      v16bf bt = load_bt(W + (size_t)(headStart + t * 16) * D_ + k0, D_, lane);
      acc[0][t] = WMMA_BF16(a0, bt, acc[0][t]);
      acc[1][t] = WMMA_BF16(a1, bt, acc[1][t]);
    }
  }

  const int n  = lane & 15;
  const int mo = (lane >> 4) * 8;

  if (region == 2) {
    // V: store transposed [b][kvh][hd][s]; 8 consecutive s per lane -> b128.
    const int kvh = headStart >> 6;
    #pragma unroll
    for (int mi = 0; mi < 2; ++mi) {
      const int m0i  = m0 + mi * 16;
      const int bidx = m0i >> 11;
      const int s0   = m0i & (S_ - 1);
      #pragma unroll
      for (int t = 0; t < 4; ++t) {
        const int hd = t * 16 + n;
        v8bf pk;
        #pragma unroll
        for (int r = 0; r < 8; ++r) pk[r] = (bf16)acc[mi][t][r];
        size_t idx = (((size_t)(bidx * KVH_ + kvh) * HD_ + hd) * S_) + (s0 + mo);
        *(v8bf*)(vtb + idx) = pk;
      }
    }
  } else {
    // Q/K: apply RoPE in-register. RoPE pairs = adjacent columns = adjacent
    // lanes in the D layout -> one shfl_xor(1) fetches the partner.
    const int head = headStart >> 6;
    const float qscale = (region == 0) ? 0.125f : 1.0f;  // 1/sqrt(HD) baked in
    #pragma unroll
    for (int mi = 0; mi < 2; ++mi) {
      const int m0i  = m0 + mi * 16;
      const int bidx = m0i >> 11;
      const int s0   = m0i & (S_ - 1);
      bf16* dst = (region == 0)
          ? qb   + (size_t)(bidx * H_   + head) * S_ * HD_
          : kbuf + (size_t)(bidx * KVH_ + head) * S_ * HD_;
      #pragma unroll
      for (int t = 0; t < 4; ++t) {
        const int hd = t * 16 + n;
        const float freq = __powf(10000.0f, -(float)(hd & ~1) / 64.0f);
        #pragma unroll
        for (int r = 0; r < 8; ++r) {
          float vv = acc[mi][t][r];
          float pv = __shfl_xor(vv, 1, 32);
          int s = s0 + mo + r;
          float ang = (float)s * freq;
          float sn, cs;
          __sincosf(ang, &sn, &cs);
          float res = (hd & 1) ? (pv * sn + vv * cs) : (vv * cs - pv * sn);
          dst[(size_t)s * HD_ + hd] = (bf16)(res * qscale);
        }
      }
    }
  }
}

// ---------------------------------------------------------------------------
// Kernel 3: flash attention. grid = (S/32, H, B), one wave per block.
// Q tile 32x64 in A-regs; stream keys in chunks of 32; each K/V fragment
// feeds two WMMAs (two query row-groups). Online softmax in registers; P
// bounced through a 2KB LDS tile for D-layout -> A-layout.
// ---------------------------------------------------------------------------
__global__ __launch_bounds__(32) void flash_kernel(
    const bf16* __restrict__ qb, const bf16* __restrict__ kbuf,
    const bf16* __restrict__ vtb, const int* __restrict__ mask,
    bf16* __restrict__ ab) {
  const int lane = threadIdx.x & 31;
  const int qt = blockIdx.x, h = blockIdx.y, b = blockIdx.z;
  const int kvh = h >> 2;   // NREP = 4
  const int n = lane & 15, mo = (lane >> 4) * 8;

  const bf16* qptr = qb   + ((size_t)(b * H_ + h) * S_ + qt * 32) * HD_;
  const bf16* kptr = kbuf + (size_t)(b * KVH_ + kvh) * S_ * HD_;
  const bf16* vptr = vtb  + (size_t)(b * KVH_ + kvh) * HD_ * S_;
  const int*  mk   = mask + b * S_;

  v16bf aq[2][2];
  #pragma unroll
  for (int mi = 0; mi < 2; ++mi) {
    aq[mi][0] = load_a(qptr + (size_t)mi * 16 * HD_ + 0,  HD_, lane);
    aq[mi][1] = load_a(qptr + (size_t)mi * 16 * HD_ + 32, HD_, lane);
  }

  float rm[2][8], rl[2][8];
  v8f o[2][4];
  #pragma unroll
  for (int mi = 0; mi < 2; ++mi) {
    #pragma unroll
    for (int r = 0; r < 8; ++r) { rm[mi][r] = -1e30f; rl[mi][r] = 0.f; }
    #pragma unroll
    for (int t = 0; t < 4; ++t) o[mi][t] = zero8();
  }

  __shared__ bf16 lp[32 * 32];
  const float NEG = -1e30f;

  for (int j0 = 0; j0 < S_; j0 += 32) {
    // ---- K fragments for this 32-key chunk (reused by both row-groups) ----
    v16bf b00 = load_bt(kptr + (size_t)(j0 + 0)  * HD_ + 0,  HD_, lane);
    v16bf b01 = load_bt(kptr + (size_t)(j0 + 0)  * HD_ + 32, HD_, lane);
    v16bf b10 = load_bt(kptr + (size_t)(j0 + 16) * HD_ + 0,  HD_, lane);
    v16bf b11 = load_bt(kptr + (size_t)(j0 + 16) * HD_ + 32, HD_, lane);
    const bool msk0 = (mk[j0 + n] == 0);
    const bool msk1 = (mk[j0 + 16 + n] == 0);

    #pragma unroll
    for (int mi = 0; mi < 2; ++mi) {
      // ---- scores: 16 queries x 32 keys (pre-scaled Q) -> 4 WMMAs ----
      v8f sc0 = zero8(), sc1 = zero8();
      sc0 = WMMA_BF16(aq[mi][0], b00, sc0);
      sc0 = WMMA_BF16(aq[mi][1], b01, sc0);
      sc1 = WMMA_BF16(aq[mi][0], b10, sc1);
      sc1 = WMMA_BF16(aq[mi][1], b11, sc1);
      // ---- attention mask (per key column = per lane) ----
      if (msk0) {
        #pragma unroll
        for (int r = 0; r < 8; ++r) sc0[r] = NEG;
      }
      if (msk1) {
        #pragma unroll
        for (int r = 0; r < 8; ++r) sc1[r] = NEG;
      }
      // ---- online softmax update ----
      float corr[8];
      #pragma unroll
      for (int r = 0; r < 8; ++r) {
        float cm = rowmax16(fmaxf(sc0[r], sc1[r]));
        float nm = fmaxf(rm[mi][r], cm);
        corr[r] = __expf(rm[mi][r] - nm);
        rm[mi][r] = nm;
        float p0 = (sc0[r] > -5e29f) ? __expf(sc0[r] - nm) : 0.f;
        float p1 = (sc1[r] > -5e29f) ? __expf(sc1[r] - nm) : 0.f;
        rl[mi][r] = rl[mi][r] * corr[r] + rowsum16(p0 + p1);
        sc0[r] = p0;
        sc1[r] = p1;
      }
      #pragma unroll
      for (int t = 0; t < 4; ++t)
        #pragma unroll
        for (int r = 0; r < 8; ++r) o[mi][t][r] *= corr[r];
      // ---- P: D-layout -> LDS (row-major 32x32) ----
      #pragma unroll
      for (int r = 0; r < 8; ++r) {
        lp[(mi * 16 + mo + r) * 32 + n]      = (bf16)sc0[r];
        lp[(mi * 16 + mo + r) * 32 + 16 + n] = (bf16)sc1[r];
      }
    }
    __syncthreads();
    v16bf ap0 = load_a(lp, 32, lane);
    v16bf ap1 = load_a(lp + 16 * 32, 32, lane);
    __syncthreads();
    // ---- O += P @ V (V transposed: B^T rows contiguous in keys) ----
    #pragma unroll
    for (int t = 0; t < 4; ++t) {
      v16bf bv = load_bt(vptr + (size_t)(t * 16) * S_ + j0, S_, lane);
      o[0][t] = WMMA_BF16(ap0, bv, o[0][t]);
      o[1][t] = WMMA_BF16(ap1, bv, o[1][t]);
    }
  }

  // ---- normalize and store [b][s][h*HD+hd] as bf16 ----
  #pragma unroll
  for (int mi = 0; mi < 2; ++mi) {
    #pragma unroll
    for (int r = 0; r < 8; ++r)
      rl[mi][r] = (rl[mi][r] > 0.f) ? 1.f / rl[mi][r] : 0.f;
    #pragma unroll
    for (int t = 0; t < 4; ++t) {
      #pragma unroll
      for (int r = 0; r < 8; ++r) {
        int s = qt * 32 + mi * 16 + mo + r;
        int col = h * HD_ + t * 16 + n;
        ab[((size_t)(b * S_ + s)) * D_ + col] = (bf16)(o[mi][t][r] * rl[mi][r]);
      }
    }
  }
}

// ---------------------------------------------------------------------------
// Kernel 4: output projection out = ab @ wo^T, fp32 output.
// grid = (128, 32); each wave 32x64 tile, K = 2048 in steps of 32.
// ---------------------------------------------------------------------------
__global__ __launch_bounds__(32) void oproj_kernel(
    const bf16* __restrict__ ab, const bf16* __restrict__ wob,
    float* __restrict__ out) {
  const int lane = threadIdx.x & 31;
  const int m0 = blockIdx.x * 32, n0 = blockIdx.y * 64;
  const bf16* A = ab + (size_t)m0 * D_;

  v8f acc[2][4];
  #pragma unroll
  for (int mi = 0; mi < 2; ++mi)
    #pragma unroll
    for (int t = 0; t < 4; ++t) acc[mi][t] = zero8();

  for (int k0 = 0; k0 < D_; k0 += 32) {
    v16bf a0 = load_a(A + k0, D_, lane);
    v16bf a1 = load_a(A + (size_t)16 * D_ + k0, D_, lane);
    #pragma unroll
    for (int t = 0; t < 4; ++t) {
      v16bf bt = load_bt(wob + (size_t)(n0 + t * 16) * D_ + k0, D_, lane);
      acc[0][t] = WMMA_BF16(a0, bt, acc[0][t]);
      acc[1][t] = WMMA_BF16(a1, bt, acc[1][t]);
    }
  }

  const int n = lane & 15, mo = (lane >> 4) * 8;
  #pragma unroll
  for (int mi = 0; mi < 2; ++mi) {
    #pragma unroll
    for (int t = 0; t < 4; ++t) {
      #pragma unroll
      for (int r = 0; r < 8; ++r) {
        out[(size_t)(m0 + mi * 16 + mo + r) * D_ + n0 + t * 16 + n] = acc[mi][t][r];
      }
    }
  }
}

// ---------------------------------------------------------------------------
extern "C" void kernel_launch(void* const* d_in, const int* in_sizes, int n_in,
                              void* d_out, int out_size, void* d_ws, size_t ws_size,
                              hipStream_t stream) {
  const float* x  = (const float*)d_in[0];
  const float* wq = (const float*)d_in[1];
  const float* wk = (const float*)d_in[2];
  const float* wv = (const float*)d_in[3];
  const float* wo = (const float*)d_in[4];
  const int* mask = (const int*)d_in[5];
  float* out = (float*)d_out;

  // bf16 workspace layout (~76 MB total)
  bf16* ws = (bf16*)d_ws;
  size_t off = 0;
  bf16* xb  = ws + off; off += (size_t)B_ * S_ * D_;         // 8M
  bf16* wqb = ws + off; off += (size_t)D_ * D_;              // 4M
  bf16* wkb = ws + off; off += (size_t)KVH_ * HD_ * D_;      // 1M
  bf16* wvb = ws + off; off += (size_t)KVH_ * HD_ * D_;      // 1M
  bf16* wob = ws + off; off += (size_t)D_ * D_;              // 4M
  bf16* qb  = ws + off; off += (size_t)B_ * H_ * S_ * HD_;   // 8M
  bf16* kbf = ws + off; off += (size_t)B_ * KVH_ * S_ * HD_; // 2M
  bf16* vtb = ws + off; off += (size_t)B_ * KVH_ * HD_ * S_; // 2M
  bf16* ab  = ws + off; off += (size_t)B_ * S_ * D_;         // 8M

  auto cvt = [&](const float* in, bf16* o, size_t nel) {
    int n4 = (int)(nel / 4);
    cvt_kernel<<<(n4 + 255) / 256, 256, 0, stream>>>(in, o, n4);
  };
  cvt(x,  xb,  (size_t)B_ * S_ * D_);
  cvt(wq, wqb, (size_t)D_ * D_);
  cvt(wk, wkb, (size_t)KVH_ * HD_ * D_);
  cvt(wv, wvb, (size_t)KVH_ * HD_ * D_);
  cvt(wo, wob, (size_t)D_ * D_);

  qkv_rope_kernel<<<dim3(128, 48, 1), 32, 0, stream>>>(xb, wqb, wkb, wvb, qb, kbf, vtb);
  flash_kernel<<<dim3(S_ / 32, H_, B_), 32, 0, stream>>>(qb, kbf, vtb, mask, ab);
  oproj_kernel<<<dim3(128, 32, 1), 32, 0, stream>>>(ab, wob, out);
}
